// DDG_Net_ambiguous_43834436223252
// MI455X (gfx1250) — matrix operand order repaired and checked by reading, not compiled, not run on hardware
//
#include <hip/hip_runtime.h>
#include <hip/hip_bf16.h>
#include <stdint.h>

// ---------------------------------------------------------------------------
// CDNA5 (gfx1250) implementation of DDG-Net forward.
// All large contractions run through v_wmma_f32_16x16x32_bf16 (bf16 in, f32 acc).
// Block tile 128x128, 8 waves (wave32), wave tile 32x64 = 2x4 WMMA fragments.
// LDS double-buffered; next tile's global loads issued before the WMMA block.
// ---------------------------------------------------------------------------

typedef __attribute__((ext_vector_type(16))) __bf16 v16bf;
typedef __attribute__((ext_vector_type(8)))  float  v8f;

#define TILE_M 128
#define TILE_N 128
#define TILE_K 32
#define LDSW   40   // padded LDS stride in halves (bank-conflict avoidance)

// pack two floats to two bf16 (RTNE) in one dword via the HW packed cvt.
// (gfx1250 assembler accepts v_cvt_pk_bf16_f32: D[15:0]=bf16(S0), D[31:16]=bf16(S1))
__device__ __forceinline__ unsigned pkbf(float a, float b) {
  unsigned r;
  asm("v_cvt_pk_bf16_f32 %0, %1, %2" : "=v"(r) : "v"(a), "v"(b));
  return r;
}

union Frag16 { v16bf v; uint4 q[2]; };

// ---------------------------------------------------------------------------
// Batched GEMM:  C[b] (+)= act( A[b] * B[b](shifted cols, kmask) + bias )
//   TRANSA=0:  A elem = A[m*lda + k*astride]   (AST1=1 -> astride==1)
//   TRANSA=1:  A elem = A[k*lda + m]
//   SHIFTED=1: B columns offset by `shift`, clamp-to-zero at edges
//   MASKED=1:  B row k scaled by kmask[b*K + k]
// M % 128 == 0, N % 128 == 0, K % 32 == 0 (true for all shapes used here).
// Staging: each thread owns (LDS row, 4 consecutive k) -> one b64 LDS store;
// global loads coalesced across lanes; load phase separated from store phase.
// ---------------------------------------------------------------------------
template <int TRANSA, int AST1, int SHIFTED, int MASKED>
__global__ __launch_bounds__(256)
void gemm_wmma(const float* __restrict__ A, const float* __restrict__ Bm,
               float* __restrict__ C,
               const float* __restrict__ bias, const float* __restrict__ kmask,
               int M, int N, int K,
               int lda, int astride, long long strideA,
               int ldb, long long strideB,
               int ldc, long long strideC,
               int shift, int beta, int act)
{
  __shared__ __align__(16) unsigned short sA[2][TILE_M * LDSW];
  __shared__ __align__(16) unsigned short sB[2][TILE_N * LDSW];

  const int tid  = threadIdx.x;
  const int wave = tid >> 5, lane = tid & 31;
  const int l15  = lane & 15, hi = lane >> 4;
  const int wm   = wave >> 1, wn = wave & 1;     // 4 x 2 wave grid
  const int m0   = blockIdx.y * TILE_M;
  const int n0   = blockIdx.x * TILE_N;
  const int bz   = blockIdx.z;

  A  += (size_t)bz * strideA;
  Bm += (size_t)bz * strideB;
  C  += (size_t)bz * strideC;
  const float* km = MASKED ? (kmask + (size_t)bz * K) : nullptr;

  // staging thread ownership: row = tid&127, kg = ((tid+it*256)>>7)*4
  // A rows are m (0..127), B rows are n (0..127): both 128 rows x 8 k-groups.
  // per-thread B column (+shift clamp computed once)
  const int bn   = tid & 127;          // owned n for B staging
  int bcol = n0 + bn + (SHIFTED ? shift : 0);
  int bcc  = min(max(bcol, 0), N - 1);
  const bool bok = (bcol == bcc);

  float ax[4][4], bx[4][4];

  auto load_tiles = [&](int k0) {
    // ---- A loads ----
#pragma unroll
    for (int it = 0; it < 4; ++it) {
      int e = tid + it * 256;
      if (!TRANSA) {
        int m = e & 127, kg = (e >> 7) * 4;
        const float* src = A + (size_t)(m0 + m) * lda + (size_t)(k0 + kg) * astride;
        if (AST1) {
          float4 v = *(const float4*)src;
          ax[it][0] = v.x; ax[it][1] = v.y; ax[it][2] = v.z; ax[it][3] = v.w;
        } else {
          ax[it][0] = src[0]; ax[it][1] = src[astride];
          ax[it][2] = src[2 * astride]; ax[it][3] = src[3 * astride];
        }
      } else {
        int m = e & 127, kg = (e >> 7) * 4;
#pragma unroll
        for (int c = 0; c < 4; ++c)
          ax[it][c] = A[(size_t)(k0 + kg + c) * lda + (m0 + m)];
      }
    }
    // ---- B loads (k-major, coalesced across lanes) ----
#pragma unroll
    for (int it = 0; it < 4; ++it) {
      int e = tid + it * 256;
      int kg = (e >> 7) * 4;
#pragma unroll
      for (int c = 0; c < 4; ++c) {
        float v = Bm[(size_t)(k0 + kg + c) * ldb + (SHIFTED ? bcc : (n0 + bn))];
        if (SHIFTED && !bok) v = 0.f;
        if (MASKED) v *= km[k0 + kg + c];
        bx[it][c] = v;
      }
    }
  };

  auto store_tiles = [&](int p) {
#pragma unroll
    for (int it = 0; it < 4; ++it) {
      int e = tid + it * 256;
      int r = e & 127, kg = (e >> 7) * 4;
      uint2 t;
      t.x = pkbf(ax[it][0], ax[it][1]);
      t.y = pkbf(ax[it][2], ax[it][3]);
      *(uint2*)(&sA[p][r * LDSW + kg]) = t;
      uint2 u;
      u.x = pkbf(bx[it][0], bx[it][1]);
      u.y = pkbf(bx[it][2], bx[it][3]);
      *(uint2*)(&sB[p][r * LDSW + kg]) = u;
    }
  };

  v8f acc[2][4];
#pragma unroll
  for (int i = 0; i < 2; ++i)
#pragma unroll
    for (int j = 0; j < 4; ++j)
#pragma unroll
      for (int r = 0; r < 8; ++r) acc[i][j][r] = 0.0f;

  // prologue: stage tile 0
  load_tiles(0);
  store_tiles(0);
  __syncthreads();

  for (int k0 = 0; k0 < K; k0 += TILE_K) {
    const int p = (k0 >> 5) & 1;
    const bool more = (k0 + TILE_K) < K;

    // fragments per ISA 7.12.2 layouts (from buffer p)
    Frag16 af[2], bfr[4];
#pragma unroll
    for (int ti = 0; ti < 2; ++ti) {
      int row = wm * 32 + ti * 16 + l15;
      af[ti].q[0] = *(const uint4*)(&sA[p][row * LDSW + hi * 8]);       // K = hi*8 + 0..7
      af[ti].q[1] = *(const uint4*)(&sA[p][row * LDSW + 16 + hi * 8]);  // K = 16 + hi*8 + 0..7
    }
#pragma unroll
    for (int tj = 0; tj < 4; ++tj) {
      int nn = wn * 64 + tj * 16 + l15;
      bfr[tj].q[0] = *(const uint4*)(&sB[p][nn * LDSW + hi * 8]);       // K = hi*8 + 0..7
      bfr[tj].q[1] = *(const uint4*)(&sB[p][nn * LDSW + 16 + hi * 8]);  // K = 16 + hi*8 + 0..7
    }

    // issue next tile's global loads BEFORE the WMMA block (latency overlap)
    if (more) {
      load_tiles(k0 + TILE_K);
      __builtin_prefetch(Bm + (size_t)(k0 + 2 * TILE_K + (tid >> 7)) * ldb + n0 + bn, 0, 1);
    }

#pragma unroll
    for (int ti = 0; ti < 2; ++ti)
#pragma unroll
      for (int tj = 0; tj < 4; ++tj)
        acc[ti][tj] = __builtin_amdgcn_wmma_f32_16x16x32_bf16(
            false, af[ti].v, false, bfr[tj].v, (short)0, acc[ti][tj], false, false);

    if (more) store_tiles(p ^ 1);
    __syncthreads();
  }

  // -------- epilogue: beta-accumulate, bias, leaky-relu, store --------
#pragma unroll
  for (int ti = 0; ti < 2; ++ti) {
#pragma unroll
    for (int tj = 0; tj < 4; ++tj) {
      int colg = n0 + wn * 64 + tj * 16 + l15;
#pragma unroll
      for (int r = 0; r < 8; ++r) {
        int rowg = m0 + wm * 32 + ti * 16 + r + hi * 8;
        float v = acc[ti][tj][r];
        if (beta) v += C[(size_t)rowg * ldc + colg];
        if (bias) v += bias[rowg];
        if (act == 1) v = (v > 0.f) ? v : 0.2f * v;
        C[(size_t)rowg * ldc + colg] = v;
      }
    }
  }
}

// ---------------------------------------------------------------------------
// Small kernels
// ---------------------------------------------------------------------------

// out[b,t] = sigmoid( dot(w[0:H], h[b,:,t]) + b0 )
__global__ void conv3_sigmoid_kernel(const float* __restrict__ w, const float* __restrict__ b0,
                                     const float* __restrict__ h, float* __restrict__ out,
                                     int H, int T)
{
  int idx = blockIdx.x * blockDim.x + threadIdx.x;
  int bb = idx / T, t = idx - bb * T;
  const float* col = h + (size_t)bb * H * T + t;
  float s = 0.f;
  for (int i = 0; i < H; ++i) s += w[i] * col[(size_t)i * T];
  s += b0[0];
  out[idx] = 1.f / (1.f + __expf(-s));
}

// column-wise L2 normalize over D
__global__ void l2norm_kernel(const float* __restrict__ x, float* __restrict__ y, int D, int T)
{
  int b = blockIdx.y;
  int t = blockIdx.x * blockDim.x + threadIdx.x;
  const float* col = x + (size_t)b * D * T + t;
  float s = 0.f;
  for (int d = 0; d < D; ++d) { float v = col[(size_t)d * T]; s += v * v; }
  float sc = 1.f / fmaxf(sqrtf(s), 1e-12f);
  float* oc = y + (size_t)b * D * T + t;
  for (int d = 0; d < D; ++d) oc[(size_t)d * T] = col[(size_t)d * T] * sc;
}

// action/background/ambiguous rows + row-masks (needs anyact/anybg per batch)
__global__ void rows_kernel(const float* __restrict__ va, const float* __restrict__ fa,
                            float* __restrict__ act, float* __restrict__ bg, float* __restrict__ amb,
                            float* __restrict__ rma, float* __restrict__ rmb, int T)
{
  int b = blockIdx.x;
  __shared__ int cA, cB;
  if (threadIdx.x == 0) { cA = 0; cB = 0; }
  __syncthreads();
  int la = 0, lb = 0;
  for (int t = threadIdx.x; t < T; t += blockDim.x) {
    float v = va[b * T + t], f = fa[b * T + t];
    float a = (v >= 0.55f && f >= 0.55f) ? 1.f : 0.f;
    float g = (v <  0.45f && f <  0.45f) ? 1.f : 0.f;
    act[b * T + t] = a; bg[b * T + t] = g; amb[b * T + t] = 1.f - a - g;
    la += (int)a; lb += (int)g;
  }
  atomicAdd(&cA, la); atomicAdd(&cB, lb);
  __syncthreads();
  bool anyA = cA > 0, anyB = cB > 0;
  for (int t = threadIdx.x; t < T; t += blockDim.x) {
    rma[b * T + t] = (bg[b * T + t]  == 0.f && anyA) ? 1.f : 0.f;
    rmb[b * T + t] = (act[b * T + t] == 0.f && anyB) ? 1.f : 0.f;
  }
}

// fusion = max(0.5*(vs+fs), 0)  in-place into vs
__global__ void fuse_kernel(float* __restrict__ vs, const float* __restrict__ fs,
                            unsigned long long n)
{
  unsigned long long i = (unsigned long long)blockIdx.x * blockDim.x + threadIdx.x;
  if (i >= n) return;
  float f = 0.5f * (vs[i] + fs[i]);
  vs[i] = (f < 0.f) ? 0.f : f;
}

__device__ __forceinline__ unsigned fkey(float f) {
  unsigned u = __float_as_uint(f);
  return (u & 0x80000000u) ? ~u : (u | 0x80000000u);
}
__device__ __forceinline__ float unkey(unsigned k) {
  unsigned u = (k & 0x80000000u) ? (k & 0x7fffffffu) : ~k;
  return __uint_as_float(u);
}

// per-column (rank+1)-th largest via 4-pass radix select in LDS. T == 2048.
__global__ void topk_kernel(const float* __restrict__ fus, float* __restrict__ thr,
                            int T, int rank)
{
  __shared__ unsigned key[2048];
  __shared__ unsigned hist[256];
  __shared__ unsigned s_prefix;
  __shared__ int s_rank;
  int b = blockIdx.x / T, j = blockIdx.x - b * T;
  const float* col = fus + (size_t)b * T * T + j;
  for (int i = threadIdx.x; i < T; i += blockDim.x) key[i] = fkey(col[(size_t)i * T]);
  if (threadIdx.x == 0) { s_prefix = 0u; s_rank = rank; }
  __syncthreads();
  for (int sh = 24; sh >= 0; sh -= 8) {
    if (threadIdx.x < 256) hist[threadIdx.x] = 0u;
    __syncthreads();
    unsigned pmask = (sh == 24) ? 0u : (0xFFFFFFFFu << (sh + 8));
    unsigned pref = s_prefix;
    for (int i = threadIdx.x; i < T; i += blockDim.x) {
      unsigned k = key[i];
      if ((k & pmask) == pref) atomicAdd(&hist[(k >> sh) & 255u], 1u);
    }
    __syncthreads();
    if (threadIdx.x == 0) {
      int r = s_rank; unsigned c = 0;
      for (int bin = 255; bin >= 0; --bin) {
        int h = (int)hist[bin];
        if (r < h) { c = (unsigned)bin; break; }
        r -= h;
      }
      s_rank = r; s_prefix = pref | (c << sh);
    }
    __syncthreads();
  }
  if (threadIdx.x == 0) thr[b * T + j] = unkey(s_prefix);
}

// fus[b,i,j] = fus >= thr[b,j] ? fus : 0
__global__ void apply_thr_kernel(float* __restrict__ fus, const float* __restrict__ thr,
                                 int T, unsigned long long n)
{
  unsigned long long i = (unsigned long long)blockIdx.x * blockDim.x + threadIdx.x;
  if (i >= n) return;
  unsigned long long TT = (unsigned long long)T * T;
  int b = (int)(i / TT);
  int j = (int)(i % (unsigned long long)T);
  float f = fus[i];
  fus[i] = (f >= thr[b * T + j]) ? f : 0.f;
}

// per-column masked sums for the three L1 normalizations
__global__ void colsum_kernel(const float* __restrict__ fus,
                              const float* __restrict__ act, const float* __restrict__ bg,
                              const float* __restrict__ amb,
                              float* __restrict__ csA, float* __restrict__ csB,
                              float* __restrict__ csM, int T)
{
  int b = blockIdx.y;
  int j = blockIdx.x * blockDim.x + threadIdx.x;
  float aj = act[b * T + j], gj = bg[b * T + j], mj = amb[b * T + j];
  const float* p = fus + (size_t)b * T * T + j;
  float s1 = 0.f, s2 = 0.f, s3 = 0.f;
  for (int i = 0; i < T; ++i) {
    float f = p[(size_t)i * T];
    float ai = act[b * T + i], gi = bg[b * T + i];
    if (gi == 0.f) s1 += f;
    if (ai == 0.f) s2 += f;
    s3 += f;
  }
  csA[b * T + j] = (aj > 0.f) ? s1 : 0.f;
  csB[b * T + j] = (gj > 0.f) ? s2 : 0.f;
  csM[b * T + j] = (mj > 0.f) ? s3 : 0.f;
}

// build adj_action / adj_bg / adj_amb (+ diag of adj_amb)
__global__ void adj_build_kernel(const float* __restrict__ fus,
                                 const float* __restrict__ act, const float* __restrict__ bg,
                                 const float* __restrict__ amb,
                                 const float* __restrict__ csA, const float* __restrict__ csB,
                                 const float* __restrict__ csM,
                                 float* __restrict__ adjA, float* __restrict__ adjB,
                                 float* __restrict__ adjM, float* __restrict__ diag,
                                 int T, unsigned long long n)
{
  unsigned long long idx = (unsigned long long)blockIdx.x * blockDim.x + threadIdx.x;
  if (idx >= n) return;
  unsigned long long TT = (unsigned long long)T * T;
  int b = (int)(idx / TT);
  unsigned long long r = idx % TT;
  int i = (int)(r / (unsigned long long)T);
  int j = (int)(r % (unsigned long long)T);
  float f = fus[idx];
  float aj = act[b * T + j], gj = bg[b * T + j], mj = amb[b * T + j];
  float ai = act[b * T + i], gi = bg[b * T + i];
  float vA = (gi == 0.f && aj > 0.f) ? f / fmaxf(csA[b * T + j], 1e-12f) : 0.f;
  float vB = (ai == 0.f && gj > 0.f) ? f / fmaxf(csB[b * T + j], 1e-12f) : 0.f;
  float vM = (mj > 0.f) ? f / fmaxf(csM[b * T + j], 1e-12f) : 0.f;
  adjA[idx] = vA; adjB[idx] = vB; adjM[idx] = vM;
  if (i == j) diag[b * T + j] = vM;
}

// GCN init: a2 + b2 + feat * diag_amb[j]
__global__ void gcn_init_kernel(const float* __restrict__ ca, const float* __restrict__ cb,
                                const float* __restrict__ feat, const float* __restrict__ diag,
                                float* __restrict__ gcn, int D, int T, unsigned long long n)
{
  unsigned long long idx = (unsigned long long)blockIdx.x * blockDim.x + threadIdx.x;
  if (idx >= n) return;
  unsigned long long DT = (unsigned long long)D * T;
  int b = (int)(idx / DT);
  int t = (int)(idx % (unsigned long long)T);
  gcn[idx] = ca[idx] + cb[idx] + feat[idx] * diag[b * T + t];
}

// new = 0.5*feat + 0.25*(avg + gcn)
__global__ void combine_kernel(const float* __restrict__ feat, const float* __restrict__ avg,
                               const float* __restrict__ gcn, float* __restrict__ out,
                               unsigned long long n)
{
  unsigned long long i = (unsigned long long)blockIdx.x * blockDim.x + threadIdx.x;
  if (i >= n) return;
  out[i] = 0.5f * feat[i] + 0.25f * (avg[i] + gcn[i]);
}

// ---------------------------------------------------------------------------
// Host orchestration
// ---------------------------------------------------------------------------
extern "C" void kernel_launch(void* const* d_in, const int* in_sizes, int n_in,
                              void* d_out, int out_size, void* d_ws, size_t ws_size,
                              hipStream_t stream)
{
  (void)in_sizes; (void)n_in; (void)out_size; (void)ws_size;
  const int Bn = 2, Dn = 1024, Tn = 2048, Hn = 512;
  const long long fS  = (long long)Dn * Tn;   // feat batch stride
  const long long hS  = (long long)Hn * Tn;   // hidden batch stride
  const long long aS  = (long long)Tn * Tn;   // adjacency batch stride
  const unsigned long long nF  = (unsigned long long)Bn * Dn * Tn;
  const unsigned long long nTT = (unsigned long long)Bn * Tn * Tn;

  const float* vfeat = (const float*)d_in[0];
  const float* ffeat = (const float*)d_in[1];
  const float* avW1 = (const float*)d_in[2];  const float* avB1 = (const float*)d_in[3];
  const float* avW2 = (const float*)d_in[4];  const float* avB2 = (const float*)d_in[5];
  const float* avW3 = (const float*)d_in[6];  const float* avB3 = (const float*)d_in[7];
  const float* afW1 = (const float*)d_in[8];  const float* afB1 = (const float*)d_in[9];
  const float* afW2 = (const float*)d_in[10]; const float* afB2 = (const float*)d_in[11];
  const float* afW3 = (const float*)d_in[12]; const float* afB3 = (const float*)d_in[13];
  const float* agvW0 = (const float*)d_in[14]; const float* agvB0 = (const float*)d_in[15];
  const float* agvW1 = (const float*)d_in[16]; const float* agvB1 = (const float*)d_in[17];
  const float* agfW0 = (const float*)d_in[18]; const float* agfB0 = (const float*)d_in[19];
  const float* agfW1 = (const float*)d_in[20]; const float* agfB1 = (const float*)d_in[21];
  const float* bgvW0 = (const float*)d_in[22]; const float* bgvB0 = (const float*)d_in[23];
  const float* bgvW1 = (const float*)d_in[24]; const float* bgvB1 = (const float*)d_in[25];
  const float* bgfW0 = (const float*)d_in[26]; const float* bgfB0 = (const float*)d_in[27];
  const float* bgfW1 = (const float*)d_in[28]; const float* bgfB1 = (const float*)d_in[29];

  // output layout: [v_atn B*T][new_v B*D*T][f_atn B*T][new_f B*D*T]
  float* out   = (float*)d_out;
  float* outVA = out;
  float* outNV = out + (size_t)Bn * Tn;
  float* outFA = outNV + (size_t)Bn * Dn * Tn;
  float* outNF = outFA + (size_t)Bn * Tn;

  // workspace layout (floats)
  float* ws = (float*)d_ws;
  size_t off = 0;
  auto take = [&](size_t nelem) { float* p = ws + off; off += nelem; return p; };
  float* XnV = take((size_t)Bn * Dn * Tn);   // later reused as AVG
  float* XnF = take((size_t)Bn * Dn * Tn);   // later reused as GCN
  float* S1  = take((size_t)Bn * Tn * Tn);   // vs -> fusion (kept)
  float* S2  = take((size_t)Bn * Tn * Tn);   // fs -> adj_action
  float* S3  = take((size_t)Bn * Tn * Tn);   // adj_bg
  float* S4  = take((size_t)Bn * Tn * Tn);   // adj_amb
  float* H1  = take((size_t)Bn * Hn * Tn);
  float* H2  = take((size_t)Bn * Hn * Tn);
  float* TMP = take((size_t)Bn * Dn * Tn);
  float* CA  = take((size_t)Bn * Dn * Tn);
  float* CB  = take((size_t)Bn * Dn * Tn);
  float* actr = take((size_t)Bn * Tn);
  float* bgr  = take((size_t)Bn * Tn);
  float* ambr = take((size_t)Bn * Tn);
  float* rma  = take((size_t)Bn * Tn);
  float* rmb  = take((size_t)Bn * Tn);
  float* csA  = take((size_t)Bn * Tn);
  float* csB  = take((size_t)Bn * Tn);
  float* csM  = take((size_t)Bn * Tn);
  float* diag = take((size_t)Bn * Tn);
  float* thr  = take((size_t)Bn * Tn);
  float* atnV = take((size_t)Bn * Tn);
  float* atnF = take((size_t)Bn * Tn);
  float* AVG = XnV; float* GCN = XnF;

  // dispatch to branch-free template instantiations
  auto gemm = [&](const float* A, const float* Bm, float* C, const float* bias,
                  const float* kmask, int M, int N, int K,
                  int lda, int astride, long long sA,
                  int ldb, long long sB, int ldc, long long sC,
                  int shift, int beta, int act, int transA) {
    dim3 g(N / TILE_N, M / TILE_M, Bn), blk(256, 1, 1);
    if (transA)
      gemm_wmma<1,1,0,0><<<g, blk, 0, stream>>>(A, Bm, C, bias, kmask, M, N, K,
          lda, astride, sA, ldb, sB, ldc, sC, shift, beta, act);
    else if (kmask)
      gemm_wmma<0,1,0,1><<<g, blk, 0, stream>>>(A, Bm, C, bias, kmask, M, N, K,
          lda, astride, sA, ldb, sB, ldc, sC, shift, beta, act);
    else if (astride != 1) {
      if (shift != 0)
        gemm_wmma<0,0,1,0><<<g, blk, 0, stream>>>(A, Bm, C, bias, kmask, M, N, K,
            lda, astride, sA, ldb, sB, ldc, sC, shift, beta, act);
      else
        gemm_wmma<0,0,0,0><<<g, blk, 0, stream>>>(A, Bm, C, bias, kmask, M, N, K,
            lda, astride, sA, ldb, sB, ldc, sC, shift, beta, act);
    } else
      gemm_wmma<0,1,0,0><<<g, blk, 0, stream>>>(A, Bm, C, bias, kmask, M, N, K,
          lda, astride, sA, ldb, sB, ldc, sC, shift, beta, act);
  };
  auto grid1 = [](unsigned long long n) { return dim3((unsigned)((n + 255ull) / 256ull)); };

  // attention: conv(D->H,k3)+LReLU ; conv(H->H,k3)+LReLU ; conv(H->1,k1)+sigmoid
  auto attention = [&](const float* feat, const float* w1, const float* b1,
                       const float* w2, const float* b2,
                       const float* w3, const float* b3, float* atn) {
    for (int s = 0; s < 3; ++s)   // k=3 conv as 3 shifted GEMMs, fused epilogue on last
      gemm(w1 + s, feat, H1, (s == 2) ? b1 : nullptr, nullptr,
           Hn, Tn, Dn, 3 * Dn, 3, 0, Tn, fS, Tn, hS, s - 1, (s > 0) ? 1 : 0, (s == 2) ? 1 : 0, 0);
    for (int s = 0; s < 3; ++s)
      gemm(w2 + s, H1, H2, (s == 2) ? b2 : nullptr, nullptr,
           Hn, Tn, Hn, 3 * Hn, 3, 0, Tn, hS, Tn, hS, s - 1, (s > 0) ? 1 : 0, (s == 2) ? 1 : 0, 0);
    conv3_sigmoid_kernel<<<dim3((Bn * Tn) / 256), 256, 0, stream>>>(w3, b3, H2, atn, Hn, Tn);
  };

  // ---- phase 1: initial attentions + masks ----
  attention(vfeat, avW1, avB1, avW2, avB2, avW3, avB3, atnV);
  attention(ffeat, afW1, afB1, afW2, afB2, afW3, afB3, atnF);
  rows_kernel<<<dim3(Bn), 256, 0, stream>>>(atnV, atnF, actr, bgr, ambr, rma, rmb, Tn);

  // ---- phase 2: similarity + fusion + top-k + adjacencies ----
  l2norm_kernel<<<dim3(Tn / 256, Bn), 256, 0, stream>>>(vfeat, XnV, Dn, Tn);
  l2norm_kernel<<<dim3(Tn / 256, Bn), 256, 0, stream>>>(ffeat, XnF, Dn, Tn);
  gemm(XnV, XnV, S1, nullptr, nullptr, Tn, Tn, Dn, Tn, 1, fS, Tn, fS, Tn, aS, 0, 0, 0, 1); // Xn^T Xn
  gemm(XnF, XnF, S2, nullptr, nullptr, Tn, Tn, Dn, Tn, 1, fS, Tn, fS, Tn, aS, 0, 0, 0, 1);
  fuse_kernel<<<grid1(nTT), 256, 0, stream>>>(S1, S2, nTT);
  topk_kernel<<<dim3(Bn * Tn), 256, 0, stream>>>(S1, thr, Tn, Tn / 4 - 1);   // rank 511 = 512th largest
  apply_thr_kernel<<<grid1(nTT), 256, 0, stream>>>(S1, thr, Tn, nTT);
  colsum_kernel<<<dim3(Tn / 256, Bn), 256, 0, stream>>>(S1, actr, bgr, ambr, csA, csB, csM, Tn);
  adj_build_kernel<<<grid1(nTT), 256, 0, stream>>>(S1, actr, bgr, ambr, csA, csB, csM,
                                                   S2, S3, S4, diag, Tn, nTT);

  // ---- phase 3: graph branches (XnV/XnF now reused as AVG/GCN) ----
  auto branch = [&](const float* feat,
                    const float* aw0, const float* ab0, const float* aw1, const float* ab1,
                    const float* bw0, const float* bb0, const float* bw1, const float* bb1,
                    float* outNew) {
    // avg = feat@adjA + feat@adjB + feat@adjAmb
    gemm(feat, S2, AVG, nullptr, nullptr, Dn, Tn, Tn, Tn, 1, fS, Tn, aS, Tn, fS, 0, 0, 0, 0);
    gemm(feat, S3, AVG, nullptr, nullptr, Dn, Tn, Tn, Tn, 1, fS, Tn, aS, Tn, fS, 0, 1, 0, 0);
    gemm(feat, S4, AVG, nullptr, nullptr, Dn, Tn, Tn, Tn, 1, fS, Tn, aS, Tn, fS, 0, 1, 0, 0);
    // a-path: two layers of (W@x + b) @ adjA with leaky-relu
    gemm(aw0, feat, TMP, ab0, nullptr, Dn, Tn, Dn, Dn, 1, 0, Tn, fS, Tn, fS, 0, 0, 0, 0);
    gemm(TMP, S2, CA, nullptr, nullptr, Dn, Tn, Tn, Tn, 1, fS, Tn, aS, Tn, fS, 0, 0, 1, 0);
    gemm(aw1, CA, TMP, ab1, nullptr, Dn, Tn, Dn, Dn, 1, 0, Tn, fS, Tn, fS, 0, 0, 0, 0);
    gemm(TMP, S2, CA, nullptr, nullptr, Dn, Tn, Tn, Tn, 1, fS, Tn, aS, Tn, fS, 0, 0, 1, 0);
    // b-path with adjB
    gemm(bw0, feat, TMP, bb0, nullptr, Dn, Tn, Dn, Dn, 1, 0, Tn, fS, Tn, fS, 0, 0, 0, 0);
    gemm(TMP, S3, CB, nullptr, nullptr, Dn, Tn, Tn, Tn, 1, fS, Tn, aS, Tn, fS, 0, 0, 1, 0);
    gemm(bw1, CB, TMP, bb1, nullptr, Dn, Tn, Dn, Dn, 1, 0, Tn, fS, Tn, fS, 0, 0, 0, 0);
    gemm(TMP, S3, CB, nullptr, nullptr, Dn, Tn, Tn, Tn, 1, fS, Tn, aS, Tn, fS, 0, 0, 1, 0);
    // gcn = a + b + a@adj_amb_act + b@adj_amb_bg + feat*diag(adj_amb)
    gcn_init_kernel<<<grid1(nF), 256, 0, stream>>>(CA, CB, feat, diag, GCN, Dn, Tn, nF);
    gemm(CA, S4, GCN, nullptr, rma, Dn, Tn, Tn, Tn, 1, fS, Tn, aS, Tn, fS, 0, 1, 0, 0);
    gemm(CB, S4, GCN, nullptr, rmb, Dn, Tn, Tn, Tn, 1, fS, Tn, aS, Tn, fS, 0, 1, 0, 0);
    combine_kernel<<<grid1(nF), 256, 0, stream>>>(feat, AVG, GCN, outNew, nF);
  };
  branch(vfeat, agvW0, agvB0, agvW1, agvB1, bgvW0, bgvB0, bgvW1, bgvB1, outNV);
  branch(ffeat, agfW0, agfB0, agfW1, agfB1, bgfW0, bgfB0, bgfW1, bgfB1, outNF);

  // ---- phase 4: final attentions on new features ----
  attention(outNV, avW1, avB1, avW2, avB2, avW3, avB3, outVA);
  attention(outNF, afW1, afB1, afW2, afB2, afW3, afB3, outFA);
}
